// RAFTCore_83786222010972
// MI455X (gfx1250) — matrix operand order repaired and checked by reading, not compiled, not run on hardware
//
#include <hip/hip_runtime.h>

// ---------------------------------------------------------------------------
// RAFT core for MI455X (gfx1250): all GEMM-like ops (convs as implicit GEMM +
// the cost-volume GEMM) run on V_WMMA_F32_16X16X32_F16 with f32 accumulation.
// wave32 everywhere; cost-volume GEMM stages tiles through LDS, with the
// A-tile moved by the Tensor Data Mover (tensor_load_to_lds + s_wait_tensorcnt)
// when the toolchain exposes the builtin.
// ---------------------------------------------------------------------------

typedef __attribute__((ext_vector_type(16))) _Float16 v16h;
typedef __attribute__((ext_vector_type(8)))  _Float16 v8h;
typedef __attribute__((ext_vector_type(8)))  float    v8f;
typedef __attribute__((ext_vector_type(4)))  unsigned u32x4;
typedef __attribute__((ext_vector_type(4)))  int      i32x4;
typedef __attribute__((ext_vector_type(8)))  int      i32x8;

#if defined(__has_builtin)
#if __has_builtin(__builtin_amdgcn_tensor_load_to_lds) && \
    __has_builtin(__builtin_amdgcn_s_wait_tensorcnt)
#define USE_TDM 1
#endif
#endif

enum { ACT_NONE = 0, ACT_RELU = 1, ACT_SIG = 2, ACT_TANH = 3, ACT_QSCL = 4 };

__device__ __forceinline__ v8f wmma16(v16h a, v16h b, v8f c) {
  return __builtin_amdgcn_wmma_f32_16x16x32_f16(false, a, false, b, (short)0, c,
                                                false, false);
}

// A-matrix fragment (16x32 f16, row-major source, row stride = Kpad halves).
// Layout: lanes 0-15 row M=lane, halves 0..15 -> K=hi*16+0..15
// (contiguous, matching the B fragments; hi = lane>=16).
__device__ __forceinline__ v16h load_afrag(const _Float16* __restrict__ p, int hi) {
  v8h lo = *reinterpret_cast<const v8h*>(p + hi * 16);
  v8h hh = *reinterpret_cast<const v8h*>(p + hi * 16 + 8);
  v16h a = {};
#pragma unroll
  for (int i = 0; i < 8; ++i) { a[i] = lo[i]; a[8 + i] = hh[i]; }
  return a;
}

// B-matrix fragment (32x16 f16): lane = column N; lanes 0-15 hold K=0..15,
// lanes 16-31 hold K=16..31 (contiguous). Source: per-column 32 K halves.
__device__ __forceinline__ v16h load_bfrag16(const _Float16* __restrict__ p, int hi) {
  v8h lo = *reinterpret_cast<const v8h*>(p + hi * 16);
  v8h hh = *reinterpret_cast<const v8h*>(p + hi * 16 + 8);
  v16h b = {};
#pragma unroll
  for (int i = 0; i < 8; ++i) { b[i] = lo[i]; b[8 + i] = hh[i]; }
  return b;
}

__device__ __forceinline__ float nan2num(float v) {
  if (__builtin_isnan(v)) return 0.f;
  if (__builtin_isinf(v)) return v > 0.f ? 3.4028234663852886e38f
                                         : -3.4028234663852886e38f;
  return v;
}

// ---------------------------------------------------------------------------
// Generic implicit-GEMM convolution. One wave computes a 64(out-ch) x
// 32(pixel) tile = 4x2 WMMA subtiles (8 v_wmma per 32-K chunk), amortizing the
// activation gather across the whole weight-tile column. int32 index math.
// Weights pre-converted to f16 [CoutPad64, Kpad] (zero padded).
// ---------------------------------------------------------------------------
template <int CIN, int COUT, int KH, int KW, int PH, int PW, int ACT>
__global__ __launch_bounds__(128) void k_conv_wmma(
    const float* __restrict__ x, const _Float16* __restrict__ w,
    const float* __restrict__ bias, float* __restrict__ y, int H_, int W_) {
  constexpr int Kreal = CIN * KH * KW;
  constexpr int Kpad = (Kreal + 31) & ~31;
  const int hw = H_ * W_;
  const int lane = threadIdx.x & 31;
  const int wave = threadIdx.x >> 5;
  const int hi = lane >> 4;
  const int ml = lane & 15;
  const int n0 = (blockIdx.x * 4 + wave) * 32;
  const int oc0 = blockIdx.y * 64;

  // This lane's pixel for B subtile 0 / subtile 1 (it is the lane's column).
  const int na = n0 + ml, nb = n0 + 16 + ml;
  const int ba = na / hw, raa = na - ba * hw;
  const int ya = raa / W_, xa = raa - ya * W_;
  const int bb = nb / hw, rbb = nb - bb * hw;
  const int yb = rbb / W_, xb = rbb - yb * W_;
  const int cba = ba * CIN * hw, cbb = bb * CIN * hw;

  v8f acc[4][2] = {};
  const _Float16* wr[4];
#pragma unroll
  for (int s = 0; s < 4; ++s) wr[s] = w + (size_t)(oc0 + s * 16 + ml) * Kpad;

  auto gatherB = [&](int cb, int yy, int xx, int k0) -> v16h {
    v16h bv = {};
#pragma unroll
    for (int i = 0; i < 16; ++i) {
      int k = k0 + hi * 16 + i;
      float v = 0.f;
      if (k < Kreal) {
        int ci = k / (KH * KW);
        int rr = k - ci * (KH * KW);
        int dy = rr / KW;
        int dx = rr - dy * KW;
        int iy = yy + dy - PH;
        int ix = xx + dx - PW;
        if (((unsigned)iy < (unsigned)H_) & ((unsigned)ix < (unsigned)W_))
          v = x[cb + ci * hw + iy * W_ + ix];
      }
      bv[i] = (_Float16)v;
    }
    return bv;
  };

  for (int k0 = 0; k0 < Kpad; k0 += 32) {
    __builtin_prefetch(wr[0] + k0 + 64, 0, 1);  // -> global_prefetch_b8
    v16h b0 = gatherB(cba, ya, xa, k0);
    v16h b1 = gatherB(cbb, yb, xb, k0);
#pragma unroll
    for (int s = 0; s < 4; ++s) {
      v16h a = load_afrag(wr[s] + k0, hi);
      acc[s][0] = wmma16(a, b0, acc[s][0]);
      acc[s][1] = wmma16(a, b1, acc[s][1]);
    }
  }

  auto store_tile = [&](v8f c, int ocbase, int cbo, int yy, int xx) {
#pragma unroll
    for (int r = 0; r < 8; ++r) {
      int oc = ocbase + r + hi * 8;
      if (oc < COUT) {
        float v = c[r] + bias[oc];
        if constexpr (ACT == ACT_RELU) v = fmaxf(v, 0.f);
        else if constexpr (ACT == ACT_SIG) v = 1.f / (1.f + __expf(-v));
        else if constexpr (ACT == ACT_TANH) v = tanhf(v);
        else if constexpr (ACT == ACT_QSCL) v = 0.25f * v;
        y[cbo + oc * hw + yy * W_ + xx] = v;
      }
    }
  };
  const int cboa = ba * COUT * hw, cbob = bb * COUT * hw;
#pragma unroll
  for (int s = 0; s < 4; ++s) {
    store_tile(acc[s][0], oc0 + s * 16, cboa, ya, xa);
    store_tile(acc[s][1], oc0 + s * 16, cbob, yb, xb);
  }
}

// ---------------------------------------------------------------------------
// Cost volume: Cv[b,i,j] = (1/16) * sum_c A[b,i,c] * Bm[b,c,j].
// 256-thread block -> 64x128 output tile, LDS-staged operands, 8 waves in a
// 2(M) x 4(N) arrangement, 4 WMMA per wave per 32-K chunk.
// A-tile (64 rows x 32 halves) is a 2D TDM descriptor: moved by the Tensor
// Data Mover when available; B-tile is transposed into LDS by the block.
// ---------------------------------------------------------------------------
__global__ __launch_bounds__(256) void k_corr_gemm(
    const _Float16* __restrict__ A,   // [B, hw, 256] row-major (f1 transposed)
    const _Float16* __restrict__ Bm,  // [B, 256, hw]
    float* __restrict__ Cv,           // [B, hw, hw]
    int hw) {
  __shared__ _Float16 sA[64 * 32];    // rows x K
  __shared__ _Float16 sBT[128 * 32];  // cols x K (transposed tile)
  const int b = blockIdx.z;
  const int i0 = blockIdx.y * 64;
  const int j0 = blockIdx.x * 128;
  const int t = threadIdx.x;
  const int lane = t & 31;
  const int wave = t >> 5;
  const int hi = lane >> 4;
  const int ml = lane & 15;
  const int wm = wave >> 2;  // 0..1
  const int wn = wave & 3;   // 0..3

  v8f c00 = {}, c01 = {}, c10 = {}, c11 = {};

  for (int k0 = 0; k0 < 256; k0 += 32) {
#ifdef USE_TDM
    if (wave == 0) {
      // D# group0: count=1, lds_addr, 57-bit global byte address, type=2.
      unsigned long long g =
          (unsigned long long)(size_t)(A + (size_t)(b * hw + i0) * 256 + k0);
      u32x4 g0 = {1u, (unsigned)(size_t)sA, (unsigned)g,
                  ((unsigned)((g >> 32) & 0x01FFFFFFull)) | 0x80000000u};
      // group1: data_size=2B(code 1); tensor_dim0=256, tensor_dim1=64,
      // tile_dim0=32, tile_dim1=64, tensor_dim0_stride=256.
      i32x8 g1 = {0x00010000, (int)(256u << 16), (int)(64u << 16),
                  (int)(32u << 16), 64, 256, 0, 0};
      i32x4 z4 = {};
#if __clang_major__ >= 23
      i32x8 z8 = {};
      __builtin_amdgcn_tensor_load_to_lds(g0, g1, z4, z4, z8, 0);
#else
      __builtin_amdgcn_tensor_load_to_lds(g0, g1, z4, z4, 0);
#endif
      __builtin_amdgcn_s_wait_tensorcnt(0);
    }
#else
    {  // fallback A tile: 64 rows x 32 halves, one b128 per thread
      int row = t >> 2, ch = t & 3;
      *reinterpret_cast<v8h*>(&sA[row * 32 + ch * 8]) =
          *reinterpret_cast<const v8h*>(
              A + (size_t)(b * hw + i0 + row) * 256 + k0 + ch * 8);
    }
#endif
#pragma unroll
    for (int rep = 0; rep < 2; ++rep) {  // B tile: 32 k-rows x 128 j, transpose
      int idx = t + rep * 256;
      int kr = idx >> 4;
      int jc = (idx & 15) * 8;
      v8h u = *reinterpret_cast<const v8h*>(
          Bm + (size_t)(b * 256 + k0 + kr) * hw + j0 + jc);
#pragma unroll
      for (int e = 0; e < 8; ++e) sBT[(jc + e) * 32 + kr] = u[e];
    }
    __syncthreads();
    v16h a0 = load_afrag(&sA[(wm * 32 + ml) * 32], hi);
    v16h a1 = load_afrag(&sA[(wm * 32 + 16 + ml) * 32], hi);
    v16h b0 = load_bfrag16(&sBT[(wn * 32 + ml) * 32], hi);
    v16h b1 = load_bfrag16(&sBT[(wn * 32 + 16 + ml) * 32], hi);
    c00 = wmma16(a0, b0, c00);
    c01 = wmma16(a0, b1, c01);
    c10 = wmma16(a1, b0, c10);
    c11 = wmma16(a1, b1, c11);
    __syncthreads();
  }

  const float s = 0.0625f;  // 1/sqrt(256)
  const int colA = j0 + wn * 32 + ml;
  const int colB = colA + 16;
#pragma unroll
  for (int r = 0; r < 8; ++r) {
    int row0 = i0 + wm * 32 + r + hi * 8;
    int row1 = row0 + 16;
    size_t base0 = (size_t)(b * hw + row0) * hw;
    size_t base1 = (size_t)(b * hw + row1) * hw;
    Cv[base0 + colA] = s * c00[r];
    Cv[base0 + colB] = s * c01[r];
    Cv[base1 + colA] = s * c10[r];
    Cv[base1 + colB] = s * c11[r];
  }
}

// --------------------------- small utility kernels -------------------------

__global__ void k_wconv(const float* __restrict__ src, _Float16* __restrict__ dst,
                        int Cout, int Kreal, int Kpad, int tot) {
  int tid = blockIdx.x * blockDim.x + threadIdx.x;
  if (tid >= tot) return;
  int oc = tid / Kpad, k = tid - oc * Kpad;
  float v = (oc < Cout && k < Kreal) ? src[(size_t)oc * Kreal + k] : 0.f;
  dst[tid] = (_Float16)v;
}

__global__ void k_f16cvt(const float* __restrict__ src, _Float16* __restrict__ dst,
                         int tot) {
  int tid = blockIdx.x * blockDim.x + threadIdx.x;
  if (tid < tot) dst[tid] = (_Float16)src[tid];
}

// A16[b, p, c] = f1[b, c, p]
__global__ void k_f1T(const float* __restrict__ f1, _Float16* __restrict__ dst,
                      int hw, int tot) {
  int tid = blockIdx.x * blockDim.x + threadIdx.x;
  if (tid >= tot) return;
  int c = tid & 255;
  int rest = tid >> 8;
  int b = rest / hw, p = rest - b * hw;
  dst[tid] = (_Float16)f1[(b * 256 + c) * hw + p];
}

__global__ void k_split_ctx(const float* __restrict__ ctx, float* __restrict__ hst,
                            float* __restrict__ xc, int hw, int tot) {
  int tid = blockIdx.x * blockDim.x + threadIdx.x;
  if (tid >= tot) return;
  int b = tid / (128 * hw), rem = tid - b * 128 * hw;
  hst[tid] = tanhf(ctx[(b * 256) * hw + rem]);
  xc[tid] = fmaxf(ctx[(b * 256 + 128) * hw + rem], 0.f);
}

__global__ void k_pool2(const float* __restrict__ in, float* __restrict__ out,
                        int N, int h, int w) {
  int tid = blockIdx.x * blockDim.x + threadIdx.x;
  int oh = h >> 1, ow = w >> 1;
  int tot = N * oh * ow;
  if (tid >= tot) return;
  int n = tid / (oh * ow), r = tid - n * oh * ow;
  int y = r / ow, x = r - y * ow;
  const float* p = in + ((size_t)n * h + 2 * y) * w + 2 * x;
  out[tid] = 0.25f * (p[0] + p[1] + p[w] + p[w + 1]);
}

__global__ void k_corr_lookup(const float* __restrict__ p0,
                              const float* __restrict__ p1,
                              const float* __restrict__ p2,
                              const float* __restrict__ p3,
                              const float* __restrict__ flow,
                              float* __restrict__ corr, int H_, int W_) {
  const int hw = H_ * W_;
  int tid = blockIdx.x * blockDim.x + threadIdx.x;
  int tot = 2 * hw * 324;
  if (tid >= tot) return;
  int ck = tid % 324;
  int rest = tid / 324;
  int b = rest / hw, n = rest - b * hw;
  int y = n / W_, x = n - y * W_;
  int lvl = ck / 81, k = ck - lvl * 81;
  // reference quirk: delta[...,0]=d[k/9] added to x, delta[...,1]=d[k%9] to y
  float dxv = (float)(k / 9) - 4.f;
  float dyv = (float)(k % 9) - 4.f;
  float cx = (float)x + flow[(b * 2 + 0) * hw + n];
  float cy = (float)y + flow[(b * 2 + 1) * hw + n];
  float scale = 1.f / (float)(1 << lvl);
  float sx = cx * scale + dxv;
  float sy = cy * scale + dyv;
  int fh = H_ >> lvl, fw = W_ >> lvl;
  const float* feat =
      (lvl == 0 ? p0 : lvl == 1 ? p1 : lvl == 2 ? p2 : p3) +
      (size_t)(b * hw + n) * fh * fw;
  float fx0 = floorf(sx), fy0 = floorf(sy);
  float tx = sx - fx0, ty = sy - fy0;
  int x0 = (int)fx0, y0 = (int)fy0;
  auto g = [&](int yi, int xi) -> float {
    if (xi < 0 || xi > fw - 1 || yi < 0 || yi > fh - 1) return 0.f;
    return feat[yi * fw + xi];
  };
  float v00 = g(y0, x0), v01 = g(y0, x0 + 1);
  float v10 = g(y0 + 1, x0), v11 = g(y0 + 1, x0 + 1);
  corr[(size_t)(b * 324 + ck) * hw + n] =
      v00 * (1.f - tx) * (1.f - ty) + v01 * tx * (1.f - ty) +
      v10 * (1.f - tx) * ty + v11 * tx * ty;
}

__global__ void k_zero(float* __restrict__ p, int tot) {
  int tid = blockIdx.x * blockDim.x + threadIdx.x;
  if (tid < tot) p[tid] = 0.f;
}

// dst[b, off+c, p] = src[b, c, p]  (src has SC channels, dst DC channels)
__global__ void k_copych(float* __restrict__ dst, int DC, int off,
                         const float* __restrict__ src, int SC, int hw, int tot) {
  int tid = blockIdx.x * blockDim.x + threadIdx.x;
  if (tid >= tot) return;
  int b = tid / (SC * hw), rem = tid - b * SC * hw;
  dst[(b * DC + off) * hw + rem] = src[tid];
}

// rhx channels 0..127 = r * h
__global__ void k_rh(const float* __restrict__ r, const float* __restrict__ h,
                     float* __restrict__ rhx, int hw, int tot) {
  int tid = blockIdx.x * blockDim.x + threadIdx.x;
  if (tid >= tot) return;
  int b = tid / (128 * hw), rem = tid - b * 128 * hw;
  rhx[(b * 384) * hw + rem] = r[tid] * h[tid];
}

__global__ void k_gru_combine(float* __restrict__ h, const float* __restrict__ z,
                              const float* __restrict__ q, int tot, int donan) {
  int tid = blockIdx.x * blockDim.x + threadIdx.x;
  if (tid >= tot) return;
  float o = (1.f - z[tid]) * h[tid] + z[tid] * q[tid];
  if (donan) o = nan2num(o);
  h[tid] = o;
}

__global__ void k_flowadd(float* __restrict__ flow, const float* __restrict__ df,
                          int tot) {
  int tid = blockIdx.x * blockDim.x + threadIdx.x;
  if (tid < tot) flow[tid] += nan2num(df[tid]);
}

// convex upsample: softmax(9) over mask, weighted 3x3 patch of 8*flow
__global__ void k_upsample(const float* __restrict__ flow,
                           const float* __restrict__ um, float* __restrict__ out,
                           int H_, int W_) {
  const int hw = H_ * W_;
  int tid = blockIdx.x * blockDim.x + threadIdx.x;
  int tot = 2 * hw * 64;
  if (tid >= tot) return;
  int sub = tid & 63;
  int i = sub >> 3, j = sub & 7;
  int rest = tid >> 6;
  int b = rest / hw, n = rest - b * hw;
  int y = n / W_, x = n - y * W_;
  float m[9], mx = -1e30f;
#pragma unroll
  for (int k = 0; k < 9; ++k) {
    m[k] = um[(size_t)(b * 576 + k * 64 + i * 8 + j) * hw + n];
    mx = fmaxf(mx, m[k]);
  }
  float sum = 0.f;
#pragma unroll
  for (int k = 0; k < 9; ++k) { m[k] = __expf(m[k] - mx); sum += m[k]; }
  float inv = 1.f / sum;
#pragma unroll
  for (int ch = 0; ch < 2; ++ch) {
    float acc = 0.f;
#pragma unroll
    for (int k = 0; k < 9; ++k) {
      int yy = y + k / 3 - 1, xx = x + k % 3 - 1;
      float f = 0.f;
      if (yy >= 0 && yy < H_ && xx >= 0 && xx < W_)
        f = 8.f * flow[(b * 2 + ch) * hw + yy * W_ + xx];
      acc += m[k] * inv * f;
    }
    out[((size_t)(b * 2 + ch) * (8 * H_) + (y * 8 + i)) * (size_t)(8 * W_) +
        (x * 8 + j)] = acc;
  }
}

// --------------------------- host-side orchestration -----------------------

template <int CIN, int COUT, int KH, int KW, int PH, int PW, int ACT>
static void launch_conv(const float* x, const _Float16* w, const float* b,
                        float* y, int H_, int W_, int Npix, hipStream_t s) {
  constexpr int CP = (COUT + 63) & ~63;
  k_conv_wmma<CIN, COUT, KH, KW, PH, PW, ACT>
      <<<dim3(Npix / 128, CP / 64), dim3(128), 0, s>>>(x, w, b, y, H_, W_);
}

static inline int nblk(long long n) { return (int)((n + 255) / 256); }

extern "C" void kernel_launch(void* const* d_in, const int* in_sizes, int n_in,
                              void* d_out, int out_size, void* d_ws,
                              size_t ws_size, hipStream_t stream) {
  (void)in_sizes; (void)n_in; (void)ws_size;
  const int B = 2, H = 56, W = 128;
  const int hw = H * W;           // 7168
  const int Npix = B * hw;        // 14336

  const float* fmap1 = (const float*)d_in[0];
  const float* fmap2 = (const float*)d_in[1];
  const float* cnet  = (const float*)d_in[2];
  // params dict is built in sorted(name) order:
  const float* ca_b  = (const float*)d_in[3];
  const float* ca_w  = (const float*)d_in[4];
  const float* fa_b  = (const float*)d_in[5];
  const float* fa_w  = (const float*)d_in[6];
  const float* fh1_b = (const float*)d_in[7];
  const float* fh1_w = (const float*)d_in[8];
  const float* fh2_b = (const float*)d_in[9];
  const float* fh2_w = (const float*)d_in[10];
  const float* gq1_b = (const float*)d_in[11];
  const float* gq1_w = (const float*)d_in[12];
  const float* gq2_b = (const float*)d_in[13];
  const float* gq2_w = (const float*)d_in[14];
  const float* gr1_b = (const float*)d_in[15];
  const float* gr1_w = (const float*)d_in[16];
  const float* gr2_b = (const float*)d_in[17];
  const float* gr2_w = (const float*)d_in[18];
  const float* gz1_b = (const float*)d_in[19];
  const float* gz1_w = (const float*)d_in[20];
  const float* gz2_b = (const float*)d_in[21];
  const float* gz2_w = (const float*)d_in[22];
  const float* m_b   = (const float*)d_in[23];
  const float* m_w   = (const float*)d_in[24];
  const float* mc1_b = (const float*)d_in[25];
  const float* mc1_w = (const float*)d_in[26];
  const float* mc2_b = (const float*)d_in[27];
  const float* mc2_w = (const float*)d_in[28];
  const float* mf1_b = (const float*)d_in[29];
  const float* mf1_w = (const float*)d_in[30];
  const float* mf2_b = (const float*)d_in[31];
  const float* mf2_w = (const float*)d_in[32];
  const float* u1_b  = (const float*)d_in[33];
  const float* u1_w  = (const float*)d_in[34];
  const float* u2_b  = (const float*)d_in[35];
  const float* u2_w  = (const float*)d_in[36];

  char* wsb = (char*)d_ws;
  size_t off = 0;
  auto alloc = [&](size_t bytes) -> void* {
    void* p = wsb + off;
    off = (off + bytes + 255) & ~(size_t)255;
    return p;
  };

  // f16 weight conversion (Cout padded to 64, K padded to 32)
  auto wcvt = [&](const float* src, int Cout, int Kreal) -> _Float16* {
    int CP = (Cout + 63) & ~63;
    int KP = (Kreal + 31) & ~31;
    _Float16* dst = (_Float16*)alloc((size_t)CP * KP * sizeof(_Float16));
    int tot = CP * KP;
    k_wconv<<<nblk(tot), 256, 0, stream>>>(src, dst, Cout, Kreal, KP, tot);
    return dst;
  };

  _Float16* w_fa  = wcvt(fa_w, 256, 128);
  _Float16* w_ca  = wcvt(ca_w, 256, 128);
  _Float16* w_mc1 = wcvt(mc1_w, 256, 324);
  _Float16* w_mc2 = wcvt(mc2_w, 192, 2304);
  _Float16* w_mf1 = wcvt(mf1_w, 128, 98);
  _Float16* w_mf2 = wcvt(mf2_w, 64, 1152);
  _Float16* w_m   = wcvt(m_w, 126, 2304);
  _Float16* w_gz1 = wcvt(gz1_w, 128, 1920);
  _Float16* w_gr1 = wcvt(gr1_w, 128, 1920);
  _Float16* w_gq1 = wcvt(gq1_w, 128, 1920);
  _Float16* w_gz2 = wcvt(gz2_w, 128, 1920);
  _Float16* w_gr2 = wcvt(gr2_w, 128, 1920);
  _Float16* w_gq2 = wcvt(gq2_w, 128, 1920);
  _Float16* w_fh1 = wcvt(fh1_w, 256, 1152);
  _Float16* w_fh2 = wcvt(fh2_w, 2, 2304);
  _Float16* w_u1  = wcvt(u1_w, 256, 1152);
  _Float16* w_u2  = wcvt(u2_w, 576, 256);

  const size_t szA256 = (size_t)B * 256 * hw * sizeof(float);
  float* f1   = (float*)alloc(szA256);
  float* f2   = (float*)alloc(szA256);
  float* ctx  = (float*)alloc(szA256);
  _Float16* A16 = (_Float16*)alloc((size_t)B * hw * 256 * sizeof(_Float16));
  _Float16* B16 = (_Float16*)alloc((size_t)B * 256 * hw * sizeof(_Float16));
  float* pyr0 = (float*)alloc((size_t)Npix * hw * sizeof(float));
  float* pyr1 = (float*)alloc((size_t)Npix * (hw / 4) * sizeof(float));
  float* pyr2 = (float*)alloc((size_t)Npix * (hw / 16) * sizeof(float));
  float* pyr3 = (float*)alloc((size_t)Npix * (hw / 64) * sizeof(float));
  float* hst  = (float*)alloc((size_t)B * 128 * hw * sizeof(float));
  float* xctx = (float*)alloc((size_t)B * 128 * hw * sizeof(float));
  float* corr = (float*)alloc((size_t)B * 324 * hw * sizeof(float));
  float* cf1  = (float*)alloc(szA256);
  float* cfb  = (float*)alloc((size_t)B * 192 * hw * sizeof(float));
  float* ff1  = (float*)alloc((size_t)B * 128 * hw * sizeof(float));
  float* ffb  = (float*)alloc((size_t)B * 64 * hw * sizeof(float));
  float* cff  = (float*)alloc(szA256);
  float* mfb  = (float*)alloc((size_t)B * 126 * hw * sizeof(float));
  float* xfull= (float*)alloc(szA256);
  float* hx   = (float*)alloc((size_t)B * 384 * hw * sizeof(float));
  float* rhx  = (float*)alloc((size_t)B * 384 * hw * sizeof(float));
  float* zb   = (float*)alloc((size_t)B * 128 * hw * sizeof(float));
  float* rb   = (float*)alloc((size_t)B * 128 * hw * sizeof(float));
  float* qb   = (float*)alloc((size_t)B * 128 * hw * sizeof(float));
  float* t1   = (float*)alloc(szA256);
  float* df   = (float*)alloc((size_t)B * 2 * hw * sizeof(float));
  float* um   = (float*)alloc((size_t)B * 576 * hw * sizeof(float));
  float* flow = (float*)alloc((size_t)B * 2 * hw * sizeof(float));

  // ---- feature / context projections (1x1 convs, WMMA) ----
  launch_conv<128, 256, 1, 1, 0, 0, ACT_NONE>(fmap1, w_fa, fa_b, f1, H, W, Npix, stream);
  launch_conv<128, 256, 1, 1, 0, 0, ACT_NONE>(fmap2, w_fa, fa_b, f2, H, W, Npix, stream);
  launch_conv<128, 256, 1, 1, 0, 0, ACT_NONE>(cnet, w_ca, ca_b, ctx, H, W, Npix, stream);
  k_split_ctx<<<nblk((long long)B * 128 * hw), 256, 0, stream>>>(ctx, hst, xctx, hw, B * 128 * hw);

  // ---- cost volume pyramid ----
  k_f1T<<<nblk((long long)B * hw * 256), 256, 0, stream>>>(f1, A16, hw, B * hw * 256);
  k_f16cvt<<<nblk((long long)B * 256 * hw), 256, 0, stream>>>(f2, B16, B * 256 * hw);
  k_corr_gemm<<<dim3(hw / 128, hw / 64, B), 256, 0, stream>>>(A16, B16, pyr0, hw);
  k_pool2<<<nblk((long long)Npix * (hw / 4)), 256, 0, stream>>>(pyr0, pyr1, Npix, H, W);
  k_pool2<<<nblk((long long)Npix * (hw / 16)), 256, 0, stream>>>(pyr1, pyr2, Npix, H / 2, W / 2);
  k_pool2<<<nblk((long long)Npix * (hw / 64)), 256, 0, stream>>>(pyr2, pyr3, Npix, H / 4, W / 4);

  k_zero<<<nblk((long long)B * 2 * hw), 256, 0, stream>>>(flow, B * 2 * hw);
  // xfull channels 0..127 = context features (constant across iterations)
  k_copych<<<nblk((long long)B * 128 * hw), 256, 0, stream>>>(xfull, 256, 0, xctx, 128, hw, B * 128 * hw);

  const int NIT = out_size / (B * 2 * (8 * H) * (8 * W));  // == 4
  for (int it = 0; it < NIT; ++it) {
    // correlation lookup (4 pyramid levels x 81 offsets, bilinear)
    k_corr_lookup<<<nblk((long long)2 * hw * 324), 256, 0, stream>>>(
        pyr0, pyr1, pyr2, pyr3, flow, corr, H, W);
    // motion encoder
    launch_conv<324, 256, 1, 1, 0, 0, ACT_RELU>(corr, w_mc1, mc1_b, cf1, H, W, Npix, stream);
    launch_conv<256, 192, 3, 3, 1, 1, ACT_RELU>(cf1, w_mc2, mc2_b, cfb, H, W, Npix, stream);
    launch_conv<2, 128, 7, 7, 3, 3, ACT_RELU>(flow, w_mf1, mf1_b, ff1, H, W, Npix, stream);
    launch_conv<128, 64, 3, 3, 1, 1, ACT_RELU>(ff1, w_mf2, mf2_b, ffb, H, W, Npix, stream);
    k_copych<<<nblk((long long)B * 192 * hw), 256, 0, stream>>>(cff, 256, 0, cfb, 192, hw, B * 192 * hw);
    k_copych<<<nblk((long long)B * 64 * hw), 256, 0, stream>>>(cff, 256, 192, ffb, 64, hw, B * 64 * hw);
    launch_conv<256, 126, 3, 3, 1, 1, ACT_RELU>(cff, w_m, m_b, mfb, H, W, Npix, stream);
    k_copych<<<nblk((long long)B * 126 * hw), 256, 0, stream>>>(xfull, 256, 128, mfb, 126, hw, B * 126 * hw);
    k_copych<<<nblk((long long)B * 2 * hw), 256, 0, stream>>>(xfull, 256, 254, flow, 2, hw, B * 2 * hw);

    // GRU half 1 (1x5 convs)
    k_copych<<<nblk((long long)B * 128 * hw), 256, 0, stream>>>(hx, 384, 0, hst, 128, hw, B * 128 * hw);
    k_copych<<<nblk((long long)B * 256 * hw), 256, 0, stream>>>(hx, 384, 128, xfull, 256, hw, B * 256 * hw);
    launch_conv<384, 128, 1, 5, 0, 2, ACT_SIG>(hx, w_gz1, gz1_b, zb, H, W, Npix, stream);
    launch_conv<384, 128, 1, 5, 0, 2, ACT_SIG>(hx, w_gr1, gr1_b, rb, H, W, Npix, stream);
    k_rh<<<nblk((long long)B * 128 * hw), 256, 0, stream>>>(rb, hst, rhx, hw, B * 128 * hw);
    k_copych<<<nblk((long long)B * 256 * hw), 256, 0, stream>>>(rhx, 384, 128, xfull, 256, hw, B * 256 * hw);
    launch_conv<384, 128, 1, 5, 0, 2, ACT_TANH>(rhx, w_gq1, gq1_b, qb, H, W, Npix, stream);
    k_gru_combine<<<nblk((long long)B * 128 * hw), 256, 0, stream>>>(hst, zb, qb, B * 128 * hw, 0);

    // GRU half 2 (5x1 convs)
    k_copych<<<nblk((long long)B * 128 * hw), 256, 0, stream>>>(hx, 384, 0, hst, 128, hw, B * 128 * hw);
    launch_conv<384, 128, 5, 1, 2, 0, ACT_SIG>(hx, w_gz2, gz2_b, zb, H, W, Npix, stream);
    launch_conv<384, 128, 5, 1, 2, 0, ACT_SIG>(hx, w_gr2, gr2_b, rb, H, W, Npix, stream);
    k_rh<<<nblk((long long)B * 128 * hw), 256, 0, stream>>>(rb, hst, rhx, hw, B * 128 * hw);
    launch_conv<384, 128, 5, 1, 2, 0, ACT_TANH>(rhx, w_gq2, gq2_b, qb, H, W, Npix, stream);
    k_gru_combine<<<nblk((long long)B * 128 * hw), 256, 0, stream>>>(hst, zb, qb, B * 128 * hw, 1);

    // flow head
    launch_conv<128, 256, 3, 3, 1, 1, ACT_RELU>(hst, w_fh1, fh1_b, t1, H, W, Npix, stream);
    launch_conv<256, 2, 3, 3, 1, 1, ACT_NONE>(t1, w_fh2, fh2_b, df, H, W, Npix, stream);
    k_flowadd<<<nblk((long long)B * 2 * hw), 256, 0, stream>>>(flow, df, B * 2 * hw);

    // upsample mask + convex upsample into this iteration's output slice
    launch_conv<128, 256, 3, 3, 1, 1, ACT_RELU>(hst, w_u1, u1_b, t1, H, W, Npix, stream);
    launch_conv<256, 576, 1, 1, 0, 0, ACT_QSCL>(t1, w_u2, u2_b, um, H, W, Npix, stream);
    float* outp = (float*)d_out + (size_t)it * B * 2 * (8 * H) * (8 * W);
    k_upsample<<<nblk((long long)2 * hw * 64), 256, 0, stream>>>(flow, um, outp, H, W);
  }
}
